// SecretRegenerator_24850680774867
// MI455X (gfx1250) — compile-verified, matrix operand order checked by reading, not run on hardware
//
#include <hip/hip_runtime.h>
#include <hip/hip_bf16.h>

// Sizes fixed by the reference.
#define SV 32000
#define SE 64
#define SH 128
#define SS 1024
#define SB 128
#define ST 64

typedef __attribute__((ext_vector_type(16))) _Float16 v16h;
typedef __attribute__((ext_vector_type(8)))  float    v8f;

// ---------------------------------------------------------------------------
// WMMA helpers (CDNA5 wave32, V_WMMA_F32_16X16X32_F16)
// 16-bit A 16x32 layout: lane = 16*half + r; element i -> k = (i/8)*16 + half*8 + (i%8)
// B (K x N) uses the same striping with lane%16 = n (column of B = row of W[n,k]).
// C/D f32: c[i] = C[i + 8*(lane/16)][lane%16]
// ---------------------------------------------------------------------------
__device__ inline v16h ldfrag(const _Float16* p, int row0, int ld, int k0) {
    int lane = threadIdx.x & 31;
    const _Float16* q = p + (size_t)(row0 + (lane & 15)) * ld + k0 + (lane >> 4) * 8;
    v16h v;
#pragma unroll
    for (int i = 0; i < 8; ++i) { v[i] = q[i]; v[i + 8] = q[i + 16]; }
    return v;
}

__device__ inline v8f wmma_f16(v16h a, v16h b, v8f c) {
    return __builtin_amdgcn_wmma_f32_16x16x32_f16(false, a, false, b, (short)0, c,
                                                  false, false);
}

__device__ inline float sigm(float x) { return 1.0f / (1.0f + __expf(-x)); }
__device__ inline float fast_tanh(float x) {
    x = fminf(fmaxf(x, -15.f), 15.f);
    float e = __expf(2.f * x);
    return (e - 1.f) / (e + 1.f);
}

// ---------------------------------------------------------------------------
// Setup kernels: fp32 -> f16 weight conversion, attn_W split, token init
// ---------------------------------------------------------------------------
__global__ void k_f32_to_f16(const float* __restrict__ src, _Float16* __restrict__ dst, int n) {
    int i = blockIdx.x * blockDim.x + threadIdx.x;
    if (i < n) dst[i] = (_Float16)src[i];
}

__global__ void k_split_attn(const float* __restrict__ attnW,
                             _Float16* __restrict__ Wh, _Float16* __restrict__ We) {
    int i = blockIdx.x * blockDim.x + threadIdx.x;   // 0..16383
    if (i < SH * SH) {
        int j = i >> 7, k = i & 127;
        Wh[i] = (_Float16)attnW[j * 256 + k];
        We[i] = (_Float16)attnW[j * 256 + 128 + k];
    }
}

__global__ void k_init_tok(const int* __restrict__ seq, int* __restrict__ tok) {
    int b = threadIdx.x;
    if (b < SB) tok[b] = seq[b];          // seq[0, b]
}

// ---------------------------------------------------------------------------
// Encoder: persistent single workgroup (32 waves), 1024 sequential GRU steps.
// ALL recurrent operands live in LDS (dynamic, ~208 KB of the WGP's 320 KB):
//   hs  [128][136] f16  hidden state        (padded: frag rows hit 16 banks)
//   xs  [128][72]  f16  current input
//   wh  [384][136] f16  Whh staged once     (kills the scratch-spill of
//   wi  [384][72]  f16  Wih staged once      loop-invariant B fragments)
// Each wave owns M-tile mt = w/4 and gate-column tiles ct in {w%4, w%4+4},
// so r/z/n accumulators for the SAME h-columns stay in registers.
// ---------------------------------------------------------------------------
#define ENC_HS   0
#define ENC_XS   (128 * 136)
#define ENC_WH   (ENC_XS + 128 * 72)
#define ENC_WI   (ENC_WH + 384 * 136)
#define ENC_LDSH (ENC_WI + 384 * 72)          // total halves
#define ENC_LDSB (ENC_LDSH * 2)               // bytes (212992)

__global__ void __launch_bounds__(1024) k_encoder(
    const int* __restrict__ seq, const float* __restrict__ emb,
    const _Float16* __restrict__ Wih, const _Float16* __restrict__ Whh,
    const float* __restrict__ bih, const float* __restrict__ bhh,
    _Float16* __restrict__ enc16, _Float16* __restrict__ h16d) {
    extern __shared__ _Float16 dsm[];
    _Float16* hs = dsm + ENC_HS;
    _Float16* xs = dsm + ENC_XS;
    _Float16* wh = dsm + ENC_WH;
    _Float16* wi = dsm + ENC_WI;

    const int tid = threadIdx.x;
    const int w = tid >> 5, lane = tid & 31;
    const int mt = w >> 2, c0 = w & 3;
    const int half = lane >> 4, ln = lane & 15;

    for (int i = tid; i < 128 * 136; i += 1024) hs[i] = (_Float16)0.f;
    {   // x for s = 0 (gather + f32->f16)
        int b = tid >> 3, e0 = (tid & 7) * 8;
        int row = seq[b];
#pragma unroll
        for (int j = 0; j < 8; ++j)
            xs[b * 72 + e0 + j] = (_Float16)emb[(size_t)row * SE + e0 + j];
    }
    // stage weights into padded LDS (once)
    for (int i = tid; i < 384 * SH; i += 1024) {
        int r = i >> 7, c = i & 127;
        wh[r * 136 + c] = Whh[i];
    }
    for (int i = tid; i < 384 * SE; i += 1024) {
        int r = i >> 6, c = i & 63;
        wi[r * 72 + c] = Wih[i];
    }
    __syncthreads();

    float bi[2][3], bh[2][3];
#pragma unroll
    for (int t = 0; t < 2; ++t) {
        int ct = c0 + 4 * t;
#pragma unroll
        for (int g = 0; g < 3; ++g) {
            int j = g * SH + ct * 16 + ln;
            bi[t][g] = bih[j];
            bh[t][g] = bhh[j];
        }
    }

    for (int s = 0; s < SS; ++s) {
        v16h ah[4], ax[2];
#pragma unroll
        for (int kk = 0; kk < 4; ++kk) ah[kk] = ldfrag(hs, mt * 16, 136, kk * 32);
#pragma unroll
        for (int kk = 0; kk < 2; ++kk) ax[kk] = ldfrag(xs, mt * 16, 72, kk * 32);
        __syncthreads();   // all reads of hs/xs captured in registers

#pragma unroll
        for (int t = 0; t < 2; ++t) {
            int ct = c0 + 4 * t;
            v8f gr = {}, gz = {}, gn = {}, ir = {}, iz = {}, inn = {};
#pragma unroll
            for (int kk = 0; kk < 4; ++kk) {
                gr = wmma_f16(ah[kk], ldfrag(wh, 0 * SH + ct * 16, 136, kk * 32), gr);
                gz = wmma_f16(ah[kk], ldfrag(wh, 1 * SH + ct * 16, 136, kk * 32), gz);
                gn = wmma_f16(ah[kk], ldfrag(wh, 2 * SH + ct * 16, 136, kk * 32), gn);
            }
#pragma unroll
            for (int kk = 0; kk < 2; ++kk) {
                ir  = wmma_f16(ax[kk], ldfrag(wi, 0 * SH + ct * 16, 72, kk * 32), ir);
                iz  = wmma_f16(ax[kk], ldfrag(wi, 1 * SH + ct * 16, 72, kk * 32), iz);
                inn = wmma_f16(ax[kk], ldfrag(wi, 2 * SH + ct * 16, 72, kk * 32), inn);
            }
#pragma unroll
            for (int i = 0; i < 8; ++i) {
                float r = sigm(ir[i] + bi[t][0] + gr[i] + bh[t][0]);
                float z = sigm(iz[i] + bi[t][1] + gz[i] + bh[t][1]);
                float n = fast_tanh(inn[i] + bi[t][2] + r * (gn[i] + bh[t][2]));
                int m = mt * 16 + i + half * 8;      // batch index
                int c = ct * 16 + ln;                // hidden index
                float hold = (float)hs[m * 136 + c];
                float h2 = (1.f - z) * n + z * hold;
                hs[m * 136 + c] = (_Float16)h2;      // own tile only
                enc16[((size_t)s * SB + m) * SH + c] = (_Float16)h2;
            }
        }
        if (s + 1 < SS) {   // stage x for next step (xs reads were pre-barrier)
            int b = tid >> 3, e0 = (tid & 7) * 8;
            int row = seq[(s + 1) * SB + b];
#pragma unroll
            for (int j = 0; j < 8; ++j)
                xs[b * 72 + e0 + j] = (_Float16)emb[(size_t)row * SE + e0 + j];
        }
        __syncthreads();
    }
    for (int i = tid; i < SB * SH; i += 1024) {
        int m = i >> 7, c = i & 127;
        h16d[i] = hs[m * 136 + c];
    }
}

// ---------------------------------------------------------------------------
// Ue = enc_out @ We^T  (once):  [S*B, 128] x [128, 128] -> f16
// ---------------------------------------------------------------------------
__global__ void __launch_bounds__(256) k_ue(const _Float16* __restrict__ enc16,
                                            const _Float16* __restrict__ We,
                                            _Float16* __restrict__ Ue) {
    int w = threadIdx.x >> 5, lane = threadIdx.x & 31;
    int half = lane >> 4, ln = lane & 15;
    int m0 = (blockIdx.x * 8 + w) * 16;
    v8f acc[8] = {};
#pragma unroll
    for (int kk = 0; kk < 4; ++kk) {
        v16h a = ldfrag(enc16, m0, SH, kk * 32);
#pragma unroll
        for (int nt = 0; nt < 8; ++nt)
            acc[nt] = wmma_f16(a, ldfrag(We, nt * 16, SH, kk * 32), acc[nt]);
    }
#pragma unroll
    for (int nt = 0; nt < 8; ++nt)
#pragma unroll
        for (int i = 0; i < 8; ++i) {
            size_t m = m0 + i + half * 8;
            Ue[m * SH + nt * 16 + ln] = (_Float16)acc[nt][i];
        }
}

// ---------------------------------------------------------------------------
// q = h @ Wh^T + attn_b  (128x128x128 WMMA), plus embedding gather of tok
// ---------------------------------------------------------------------------
__global__ void __launch_bounds__(256) k_q_emb(
    const _Float16* __restrict__ h16d, const _Float16* __restrict__ Wh,
    const float* __restrict__ attn_b, float* __restrict__ q,
    const int* __restrict__ tok, const float* __restrict__ emb,
    _Float16* __restrict__ xin) {
    int tid = threadIdx.x;
    int w = tid >> 5, lane = tid & 31, half = lane >> 4, ln = lane & 15;
    int m0 = w * 16;
    v8f acc[8] = {};
#pragma unroll
    for (int kk = 0; kk < 4; ++kk) {
        v16h a = ldfrag(h16d, m0, SH, kk * 32);
#pragma unroll
        for (int nt = 0; nt < 8; ++nt)
            acc[nt] = wmma_f16(a, ldfrag(Wh, nt * 16, SH, kk * 32), acc[nt]);
    }
#pragma unroll
    for (int nt = 0; nt < 8; ++nt) {
        float bb = attn_b[nt * 16 + ln];
#pragma unroll
        for (int i = 0; i < 8; ++i) {
            int m = m0 + i + half * 8;
            q[m * SH + nt * 16 + ln] = acc[nt][i] + bb;
        }
    }
    for (int i = tid; i < SB * SE; i += 256) {
        int b = i >> 6, e = i & 63;
        xin[b * 192 + e] = (_Float16)emb[(size_t)tok[b] * SE + e];
    }
}

// ---------------------------------------------------------------------------
// scores[b,s] = v . tanh(q[b] + Ue[s,b])  — one (s,b) pair per wave
// ---------------------------------------------------------------------------
__global__ void __launch_bounds__(256) k_scores(const float* __restrict__ q,
                                                const _Float16* __restrict__ Ue,
                                                const float* __restrict__ attn_v,
                                                float* __restrict__ scores) {
    int w = threadIdx.x >> 5, lane = threadIdx.x & 31;
    int idx = blockIdx.x * 8 + w;          // s*128 + b
    int s = idx >> 7, b = idx & 127;
    const _Float16* u = Ue + (size_t)idx * SH;
    const float* qb = q + b * SH;
    float sum = 0.f;
#pragma unroll
    for (int j = 0; j < 4; ++j) {
        int h = lane + 32 * j;
        sum += attn_v[h] * fast_tanh(qb[h] + (float)u[h]);
    }
#pragma unroll
    for (int off = 16; off > 0; off >>= 1) sum += __shfl_down(sum, off, 32);
    if (lane == 0) scores[b * SS + s] = sum;
}

// ---------------------------------------------------------------------------
// softmax over S per batch row + context = sum_s w[s]*enc_out[s,b,:]
// writes context into xin[:,64:] (dec GRU input) and hc[:,128:] (logits input)
// ---------------------------------------------------------------------------
__global__ void __launch_bounds__(256) k_softmax_ctx(const float* __restrict__ scores,
                                                     const _Float16* __restrict__ enc16,
                                                     _Float16* __restrict__ xin,
                                                     _Float16* __restrict__ hc) {
    __shared__ float sc[SS];
    __shared__ float red[8];
    __shared__ float ctx2[256];
    int b = blockIdx.x, tid = threadIdx.x, lane = tid & 31;
    for (int s = tid; s < SS; s += 256) sc[s] = scores[b * SS + s];
    __syncthreads();
    float m = -1e30f;
    for (int s = tid; s < SS; s += 256) m = fmaxf(m, sc[s]);
#pragma unroll
    for (int off = 16; off > 0; off >>= 1) m = fmaxf(m, __shfl_down(m, off, 32));
    if (lane == 0) red[tid >> 5] = m;
    __syncthreads();
    if (tid == 0) { float v = red[0]; for (int i = 1; i < 8; ++i) v = fmaxf(v, red[i]); red[0] = v; }
    __syncthreads();
    m = red[0];
    float sum = 0.f;
    for (int s = tid; s < SS; s += 256) { float e = __expf(sc[s] - m); sc[s] = e; sum += e; }
#pragma unroll
    for (int off = 16; off > 0; off >>= 1) sum += __shfl_down(sum, off, 32);
    __syncthreads();
    if (lane == 0) red[tid >> 5] = sum;
    __syncthreads();
    if (tid == 0) { float v = 0.f; for (int i = 0; i < 8; ++i) v += red[i]; red[0] = v; }
    __syncthreads();
    float inv = 1.0f / red[0];
    int h = tid & 127, sh = tid >> 7;
    float acc = 0.f;
    for (int s = sh * 512; s < sh * 512 + 512; ++s)
        acc += sc[s] * (float)enc16[((size_t)s * SB + b) * SH + h];
    ctx2[tid] = acc;
    __syncthreads();
    if (tid < 128) {
        float c = (ctx2[tid] + ctx2[tid + 128]) * inv;
        xin[b * 192 + 64 + h] = (_Float16)c;
        hc[b * 256 + 128 + h] = (_Float16)c;
    }
}

// ---------------------------------------------------------------------------
// Decoder GRU cell (single workgroup, same tiling as encoder, one step)
// ---------------------------------------------------------------------------
__global__ void __launch_bounds__(1024) k_dec_gru(
    const _Float16* __restrict__ xin, _Float16* __restrict__ h16d,
    const _Float16* __restrict__ dWih, const _Float16* __restrict__ dWhh,
    const float* __restrict__ bih, const float* __restrict__ bhh,
    _Float16* __restrict__ hc) {
    int tid = threadIdx.x, w = tid >> 5, lane = tid & 31;
    int half = lane >> 4, ln = lane & 15;
    int mt = w >> 2, c0 = w & 3;

    v16h ah[4], ax[6];
#pragma unroll
    for (int kk = 0; kk < 4; ++kk) ah[kk] = ldfrag(h16d, mt * 16, SH, kk * 32);
#pragma unroll
    for (int kk = 0; kk < 6; ++kk) ax[kk] = ldfrag(xin, mt * 16, 192, kk * 32);
    __syncthreads();   // all h16d reads done before anyone rewrites it

#pragma unroll
    for (int t = 0; t < 2; ++t) {
        int ct = c0 + 4 * t;
        float bi0 = bih[0 * SH + ct * 16 + ln], bh0 = bhh[0 * SH + ct * 16 + ln];
        float bi1 = bih[1 * SH + ct * 16 + ln], bh1 = bhh[1 * SH + ct * 16 + ln];
        float bi2 = bih[2 * SH + ct * 16 + ln], bh2 = bhh[2 * SH + ct * 16 + ln];
        v8f gr = {}, gz = {}, gn = {}, ir = {}, iz = {}, inn = {};
#pragma unroll
        for (int kk = 0; kk < 4; ++kk) {
            gr = wmma_f16(ah[kk], ldfrag(dWhh, 0 * SH + ct * 16, SH, kk * 32), gr);
            gz = wmma_f16(ah[kk], ldfrag(dWhh, 1 * SH + ct * 16, SH, kk * 32), gz);
            gn = wmma_f16(ah[kk], ldfrag(dWhh, 2 * SH + ct * 16, SH, kk * 32), gn);
        }
#pragma unroll
        for (int kk = 0; kk < 6; ++kk) {
            ir  = wmma_f16(ax[kk], ldfrag(dWih, 0 * SH + ct * 16, 192, kk * 32), ir);
            iz  = wmma_f16(ax[kk], ldfrag(dWih, 1 * SH + ct * 16, 192, kk * 32), iz);
            inn = wmma_f16(ax[kk], ldfrag(dWih, 2 * SH + ct * 16, 192, kk * 32), inn);
        }
#pragma unroll
        for (int i = 0; i < 8; ++i) {
            float r = sigm(ir[i] + bi0 + gr[i] + bh0);
            float z = sigm(iz[i] + bi1 + gz[i] + bh1);
            float n = fast_tanh(inn[i] + bi2 + r * (gn[i] + bh2));
            int m = mt * 16 + i + half * 8;
            int c = ct * 16 + ln;
            float hold = (float)h16d[m * SH + c];   // own tile
            float h2 = (1.f - z) * n + z * hold;
            h16d[m * SH + c] = (_Float16)h2;
            hc[m * 256 + c] = (_Float16)h2;         // logits input, first half
        }
    }
}

// ---------------------------------------------------------------------------
// logits = [h2, ctx] @ out_W^T + out_b : M=128, N=32000, K=256 (the big one)
// one wave per 16-wide N-tile, all 8 M-tiles; out_W f16 stays L2-resident
// ---------------------------------------------------------------------------
__global__ void __launch_bounds__(256) k_logits(const _Float16* __restrict__ hc,
                                                const _Float16* __restrict__ oW,
                                                const float* __restrict__ out_b,
                                                float* __restrict__ out) {
    int w = threadIdx.x >> 5, lane = threadIdx.x & 31;
    int half = lane >> 4, ln = lane & 15;
    int nt = blockIdx.x * 8 + w;           // 0..1999
    int n0 = nt * 16;
    if (nt + 1 < 2000)
        __builtin_prefetch(oW + (size_t)(n0 + 16) * 256, 0, 1);
    v8f acc[8] = {};
#pragma unroll
    for (int kk = 0; kk < 8; ++kk) {
        v16h bf = ldfrag(oW, n0, 256, kk * 32);
#pragma unroll
        for (int mt = 0; mt < 8; ++mt)
            acc[mt] = wmma_f16(ldfrag(hc, mt * 16, 256, kk * 32), bf, acc[mt]);
    }
    float bb = out_b[n0 + ln];
#pragma unroll
    for (int mt = 0; mt < 8; ++mt)
#pragma unroll
        for (int i = 0; i < 8; ++i) {
            int b = mt * 16 + i + half * 8;
            out[(size_t)b * SV + n0 + ln] = acc[mt][i] + bb;
        }
}

// ---------------------------------------------------------------------------
// argmax over V per batch row -> next token
// ---------------------------------------------------------------------------
__global__ void __launch_bounds__(256) k_argmax(const float* __restrict__ out,
                                                int* __restrict__ tok) {
    __shared__ float bv[256];
    __shared__ int bidx[256];
    int b = blockIdx.x, tid = threadIdx.x;
    const float* row = out + (size_t)b * SV;
    float best = -1e30f;
    int bi = 0;
    for (int v = tid; v < SV; v += 256) {
        float x = row[v];
        if (x > best) { best = x; bi = v; }
    }
    bv[tid] = best; bidx[tid] = bi;
    __syncthreads();
    for (int off = 128; off > 0; off >>= 1) {
        if (tid < off) {
            if (bv[tid + off] > bv[tid] ||
                (bv[tid + off] == bv[tid] && bidx[tid + off] < bidx[tid])) {
                bv[tid] = bv[tid + off];
                bidx[tid] = bidx[tid + off];
            }
        }
        __syncthreads();
    }
    if (tid == 0) tok[b] = bidx[0];
}

// ---------------------------------------------------------------------------
extern "C" void kernel_launch(void* const* d_in, const int* in_sizes, int n_in,
                              void* d_out, int out_size, void* d_ws, size_t ws_size,
                              hipStream_t stream) {
    const int*   seq     = (const int*)  d_in[0];
    const float* emb     = (const float*)d_in[2];
    const float* encWih  = (const float*)d_in[3];
    const float* encWhh  = (const float*)d_in[4];
    const float* enc_bih = (const float*)d_in[5];
    const float* enc_bhh = (const float*)d_in[6];
    const float* attnW   = (const float*)d_in[7];
    const float* attn_b  = (const float*)d_in[8];
    const float* attn_v  = (const float*)d_in[9];
    const float* decWih  = (const float*)d_in[10];
    const float* decWhh  = (const float*)d_in[11];
    const float* dec_bih = (const float*)d_in[12];
    const float* dec_bhh = (const float*)d_in[13];
    const float* outW    = (const float*)d_in[14];
    const float* out_b   = (const float*)d_in[15];
    float* out = (float*)d_out;

    // workspace carve-out (~85 MB total)
    char* base = (char*)d_ws;
    size_t off = 0;
    auto take = [&](size_t bytes) -> char* {
        char* p = base + off;
        off = (off + bytes + 255) & ~(size_t)255;
        return p;
    };
    _Float16* Wih16  = (_Float16*)take((size_t)384 * SE * 2);
    _Float16* Whh16  = (_Float16*)take((size_t)384 * SH * 2);
    _Float16* dWih16 = (_Float16*)take((size_t)384 * 192 * 2);
    _Float16* dWhh16 = (_Float16*)take((size_t)384 * SH * 2);
    _Float16* Wh16   = (_Float16*)take((size_t)SH * SH * 2);
    _Float16* We16   = (_Float16*)take((size_t)SH * SH * 2);
    _Float16* oW16   = (_Float16*)take((size_t)SV * 256 * 2);
    _Float16* enc16  = (_Float16*)take((size_t)SS * SB * SH * 2);
    _Float16* Ue16   = (_Float16*)take((size_t)SS * SB * SH * 2);
    _Float16* h16d   = (_Float16*)take((size_t)SB * SH * 2);
    _Float16* xin16  = (_Float16*)take((size_t)SB * 192 * 2);
    _Float16* hc16   = (_Float16*)take((size_t)SB * 256 * 2);
    float* q      = (float*)take((size_t)SB * SH * 4);
    float* scores = (float*)take((size_t)SB * SS * 4);
    int* tok      = (int*)take((size_t)SB * 4);

    auto cvt = [&](const float* s, _Float16* d, int n) {
        k_f32_to_f16<<<(n + 255) / 256, 256, 0, stream>>>(s, d, n);
    };
    cvt(encWih, Wih16, 384 * SE);
    cvt(encWhh, Whh16, 384 * SH);
    cvt(decWih, dWih16, 384 * 192);
    cvt(decWhh, dWhh16, 384 * SH);
    cvt(outW, oW16, SV * 256);
    k_split_attn<<<64, 256, 0, stream>>>(attnW, Wh16, We16);
    k_init_tok<<<1, 128, 0, stream>>>(seq, tok);

    // dynamic LDS: hs + xs + staged Whh/Wih (padded), ~208 KB of the 320 KB WGP LDS
    k_encoder<<<1, 1024, ENC_LDSB, stream>>>(seq, emb, Wih16, Whh16,
                                             enc_bih, enc_bhh, enc16, h16d);
    k_ue<<<1024, 256, 0, stream>>>(enc16, We16, Ue16);

    for (int t = 0; t < ST; ++t) {
        float* outt = out + (size_t)t * SB * SV;
        k_q_emb<<<1, 256, 0, stream>>>(h16d, Wh16, attn_b, q, tok, emb, xin16);
        k_scores<<<(SS * SB) / 8, 256, 0, stream>>>(q, Ue16, attn_v, scores);
        k_softmax_ctx<<<SB, 256, 0, stream>>>(scores, enc16, xin16, hc16);
        k_dec_gru<<<1, 1024, 0, stream>>>(xin16, h16d, dWih16, dWhh16,
                                          dec_bih, dec_bhh, hc16);
        k_logits<<<(SV / 16) / 8, 256, 0, stream>>>(hc16, oW16, out_b, outt);
        k_argmax<<<SB, 256, 0, stream>>>(outt, tok);
    }
    (void)in_sizes; (void)n_in; (void)out_size; (void)ws_size;
}
